// AttentionLayer_66975720013911
// MI455X (gfx1250) — matrix operand order repaired and checked by reading, not compiled, not run on hardware
//
#include <hip/hip_runtime.h>

typedef __attribute__((ext_vector_type(16))) __bf16 v16bf;
typedef __attribute__((ext_vector_type(8)))  __bf16 v8bf;
typedef __attribute__((ext_vector_type(4)))  __bf16 v4bf;
typedef __attribute__((ext_vector_type(8)))  float  v8f;

#define S_LEN   2048
#define D_HEAD  128
#define N_HEADS 16
#define N_BATCH 2
#define WIN     256
#define QTILE   16
#define NEG_BIG (-1e30f)
#define SCALE   0.08838834764831845f   // 1/sqrt(128)

__device__ __forceinline__ float half16_reduce_max(float x) {
  #pragma unroll
  for (int m = 1; m < 16; m <<= 1) x = fmaxf(x, __shfl_xor(x, m, 32));
  return x;
}
__device__ __forceinline__ float half16_reduce_sum(float x) {
  #pragma unroll
  for (int m = 1; m < 16; m <<= 1) x += __shfl_xor(x, m, 32);
  return x;
}
__device__ __forceinline__ v16bf combine(v8bf a, v8bf b) {
  return __builtin_shufflevector(a, b, 0,1,2,3,4,5,6,7,8,9,10,11,12,13,14,15);
}

// ---------------- pre-pass 1: Q (pre-scaled) and K -> bf16, same layout ----------------
__global__ __launch_bounds__(256)
void convert_qk(const float* __restrict__ Q, const float* __restrict__ K,
                __bf16* __restrict__ Qb, __bf16* __restrict__ Kb)
{
  const size_t n4 = (size_t)N_BATCH * N_HEADS * S_LEN * D_HEAD / 4;
  for (size_t i = (size_t)blockIdx.x * 256 + threadIdx.x; i < n4;
       i += (size_t)gridDim.x * 256) {
    float4 q = ((const float4*)Q)[i];
    float4 k = ((const float4*)K)[i];
    v4bf qo = { (__bf16)(q.x * SCALE), (__bf16)(q.y * SCALE),
                (__bf16)(q.z * SCALE), (__bf16)(q.w * SCALE) };
    v4bf ko = { (__bf16)k.x, (__bf16)k.y, (__bf16)k.z, (__bf16)k.w };
    ((v4bf*)Qb)[i] = qo;
    ((v4bf*)Kb)[i] = ko;
  }
}

// ---------------- pre-pass 2: V [bh,s,d] f32 -> Vt [bh,d,s] bf16 ----------------
__global__ __launch_bounds__(256)
void transpose_v(const float* __restrict__ V, __bf16* __restrict__ Vt)
{
  __shared__ __bf16 t[32][D_HEAD + 8];
  const int bh  = blockIdx.y;
  const int s0  = blockIdx.x * 32;
  const int tid = threadIdx.x;
  const int c4  = (tid & 31) * 4;   // dim base
  const int r8  = tid >> 5;         // 0..7
  const float* vb = V + (size_t)bh * S_LEN * D_HEAD;
  #pragma unroll
  for (int rr = 0; rr < 32; rr += 8) {
    const int r = rr + r8;
    float4 x = *(const float4*)(vb + (size_t)(s0 + r) * D_HEAD + c4);
    t[r][c4 + 0] = (__bf16)x.x; t[r][c4 + 1] = (__bf16)x.y;
    t[r][c4 + 2] = (__bf16)x.z; t[r][c4 + 3] = (__bf16)x.w;
  }
  __syncthreads();
  const int d  = tid >> 1;          // 0..127
  const int kh = (tid & 1) * 16;    // 0 or 16
  __bf16* orow = Vt + ((size_t)bh * D_HEAD + d) * S_LEN + s0 + kh;
  v8bf o0, o1;
  #pragma unroll
  for (int j = 0; j < 8; ++j) o0[j] = t[kh + j][d];
  #pragma unroll
  for (int j = 0; j < 8; ++j) o1[j] = t[kh + 8 + j][d];
  *(v8bf*)(orow)     = o0;
  *(v8bf*)(orow + 8) = o1;
}

// ---------------- main flash-attention kernel, bf16 operands pre-staged ----------------
__global__ __launch_bounds__(128, 1)
void fattn_bf16(const __bf16* __restrict__ Qb, const __bf16* __restrict__ Kb,
                const __bf16* __restrict__ Vt, const int* __restrict__ layer_idx,
                float* __restrict__ Out)
{
  // bf16 P staging, row stride 40 (80 B -> each 8-elem run is 16 B aligned)
  __shared__ __bf16 ps_all[4][QTILE * 40];

  const int lane = threadIdx.x & 31;
  const int wave = threadIdx.x >> 5;
  const int tile = blockIdx.x * 4 + wave;
  const int qt   = tile & (S_LEN / QTILE - 1);
  const int bh   = tile >> 7;
  const int h    = bh & (N_HEADS - 1);
  const int b    = bh >> 4;
  const int m0   = qt * QTILE;

  const bool local = (layer_idx[0] & 1) != 0;

  const int  ln    = lane & 15;
  const bool hi    = lane >= 16;
  const int  mbase = hi ? 8 : 0;
  const int  ka    = hi ? 8 : 0;    // A k base is 2*ka (16 for lanes 16-31)
  const int  kbB   = hi ? 16 : 0;   // B-layout k-row base

  const size_t head = (size_t)bh * S_LEN * D_HEAD;
  const __bf16* qbh = Qb + head;
  const __bf16* kbh = Kb + head;
  const __bf16* vth = Vt + head;    // [d][s]

  // Q tile as 4 bf16 A-fragments (two 16 B runs per chunk)
  v16bf aQ[4];
  {
    const __bf16* qrow = qbh + (size_t)(m0 + ln) * D_HEAD;
    #pragma unroll
    for (int c = 0; c < 4; ++c)
      aQ[c] = combine(*(const v8bf*)(qrow + c * 32 + 2 * ka),
                      *(const v8bf*)(qrow + c * 32 + 2 * ka + 8));
  }

  float mrow[8], lrow[8];
  v8f   Oacc[8] = {};
  #pragma unroll
  for (int r = 0; r < 8; ++r) { mrow[r] = NEG_BIG; lrow[r] = 0.f; }

  int kt_lo = 0;
  if (local) {
    int t = m0 - (WIN - 1);
    if (t < 0) t = 0;
    kt_lo = t & ~31;
  }
  __bf16* psb = &ps_all[wave][0];

  for (int kt = kt_lo; kt <= m0 + QTILE - 1; kt += 32) {
    // ---- scores (Q pre-scaled by 1/sqrt(d)) ----
    v8f sc0 = {}, sc1 = {};
    #pragma unroll
    for (int c = 0; c < 4; ++c) {
      const __bf16* kr0 = kbh + (size_t)(kt + ln)      * D_HEAD + c * 32 + kbB;
      const __bf16* kr1 = kbh + (size_t)(kt + 16 + ln) * D_HEAD + c * 32 + kbB;
      v16bf bK0 = combine(*(const v8bf*)kr0, *(const v8bf*)(kr0 + 8));
      v16bf bK1 = combine(*(const v8bf*)kr1, *(const v8bf*)(kr1 + 8));
      sc0 = __builtin_amdgcn_wmma_f32_16x16x32_bf16(false, aQ[c], false, bK0,
                                                    (short)0, sc0, false, false);
      sc1 = __builtin_amdgcn_wmma_f32_16x16x32_bf16(false, aQ[c], false, bK1,
                                                    (short)0, sc1, false, false);
    }

    // ---- mask + online softmax; stage P (bf16) into LDS ----
    #pragma unroll
    for (int r = 0; r < 8; ++r) {
      const int mi = m0 + mbase + r;
      const int j0 = kt + ln;
      const int j1 = kt + 16 + ln;
      const bool ok0 = (j0 <= mi) && (!local || (j0 > mi - WIN));
      const bool ok1 = (j1 <= mi) && (!local || (j1 > mi - WIN));
      float s0 = ok0 ? sc0[r] : NEG_BIG;
      float s1 = ok1 ? sc1[r] : NEG_BIG;
      float tmax = half16_reduce_max(fmaxf(s0, s1));
      float mnew = fmaxf(mrow[r], tmax);
      float corr = __expf(mrow[r] - mnew);
      float p0 = ok0 ? __expf(s0 - mnew) : 0.f;
      float p1 = ok1 ? __expf(s1 - mnew) : 0.f;
      lrow[r] = lrow[r] * corr + half16_reduce_sum(p0 + p1);
      mrow[r] = mnew;
      #pragma unroll
      for (int cd = 0; cd < 8; ++cd) Oacc[cd][r] *= corr;
      psb[(mbase + r) * 40 + ln]      = (__bf16)p0;
      psb[(mbase + r) * 40 + 16 + ln] = (__bf16)p1;
    }

    asm volatile("" ::: "memory");   // same-wave DS ops are in-order

    // ---- reload P as A-fragment: two aligned ds_load_b128 ----
    const __bf16* pr = psb + ln * 40;
    v16bf pA = combine(*(const v8bf*)(pr + 2 * ka),
                       *(const v8bf*)(pr + 2 * ka + 8));

    // ---- O += P V from transposed bf16 V ----
    #pragma unroll
    for (int cd = 0; cd < 8; ++cd) {
      const __bf16* vr = vth + (size_t)(cd * 16 + ln) * S_LEN + kt + kbB;
      v16bf bV = combine(*(const v8bf*)vr, *(const v8bf*)(vr + 8));
      Oacc[cd] = __builtin_amdgcn_wmma_f32_16x16x32_bf16(false, pA, false, bV,
                                                         (short)0, Oacc[cd], false, false);
    }
  }

  #pragma unroll
  for (int r = 0; r < 8; ++r) {
    const int row = m0 + mbase + r;
    const float inv = 1.0f / lrow[r];
    float* orow = Out + ((size_t)b * S_LEN + row) * (N_HEADS * D_HEAD) + h * D_HEAD + ln;
    #pragma unroll
    for (int cd = 0; cd < 8; ++cd) orow[cd * 16] = Oacc[cd][r] * inv;
  }
}

// ---------------- fallback (no workspace): converts in-loop ----------------
__global__ __launch_bounds__(128, 1)
void fattn_fallback(const float* __restrict__ Q, const float* __restrict__ K,
                    const float* __restrict__ V, const int* __restrict__ layer_idx,
                    float* __restrict__ Out)
{
  __shared__ float ps_all[4][QTILE * 33];
  const int lane = threadIdx.x & 31;
  const int wave = threadIdx.x >> 5;
  const int tile = blockIdx.x * 4 + wave;
  const int qt   = tile & (S_LEN / QTILE - 1);
  const int bh   = tile >> 7;
  const int h    = bh & (N_HEADS - 1);
  const int b    = bh >> 4;
  const int m0   = qt * QTILE;
  const bool local = (layer_idx[0] & 1) != 0;
  const int  ln    = lane & 15;
  const bool hi    = lane >= 16;
  const int  mbase = hi ? 8 : 0;
  const size_t head = (size_t)(b * N_HEADS + h) * S_LEN * D_HEAD;
  const float* qb = Q + head; const float* kb = K + head; const float* vb = V + head;

  v16bf aQ[4];
  {
    const float* qrow = qb + (size_t)(m0 + ln) * D_HEAD;
    const int ka = hi ? 8 : 0;
    #pragma unroll
    for (int c = 0; c < 4; ++c) {
      const float* p0 = qrow + c * 32 + 2 * ka;
      #pragma unroll
      for (int e = 0; e < 8; ++e) {
        aQ[c][e]     = (__bf16)(p0[e] * SCALE);
        aQ[c][8 + e] = (__bf16)(p0[8 + e] * SCALE);
      }
    }
  }
  float mrow[8], lrow[8];
  v8f Oacc[8] = {};
  #pragma unroll
  for (int r = 0; r < 8; ++r) { mrow[r] = NEG_BIG; lrow[r] = 0.f; }
  int kt_lo = 0;
  if (local) { int t = m0 - (WIN - 1); if (t < 0) t = 0; kt_lo = t & ~31; }
  float* ps = &ps_all[wave][0];
  const int kbB = hi ? 16 : 0;

  for (int kt = kt_lo; kt <= m0 + QTILE - 1; kt += 32) {
    v8f sc0 = {}, sc1 = {};
    #pragma unroll
    for (int c = 0; c < 4; ++c) {
      v16bf bK0, bK1;
      const float* kr0 = kb + (size_t)(kt + ln)      * D_HEAD + c * 32 + kbB;
      const float* kr1 = kb + (size_t)(kt + 16 + ln) * D_HEAD + c * 32 + kbB;
      #pragma unroll
      for (int e = 0; e < 16; ++e) { bK0[e] = (__bf16)kr0[e]; bK1[e] = (__bf16)kr1[e]; }
      sc0 = __builtin_amdgcn_wmma_f32_16x16x32_bf16(false, aQ[c], false, bK0,
                                                    (short)0, sc0, false, false);
      sc1 = __builtin_amdgcn_wmma_f32_16x16x32_bf16(false, aQ[c], false, bK1,
                                                    (short)0, sc1, false, false);
    }
    #pragma unroll
    for (int r = 0; r < 8; ++r) {
      const int mi = m0 + mbase + r;
      const int j0 = kt + ln, j1 = kt + 16 + ln;
      const bool ok0 = (j0 <= mi) && (!local || (j0 > mi - WIN));
      const bool ok1 = (j1 <= mi) && (!local || (j1 > mi - WIN));
      float s0 = ok0 ? sc0[r] : NEG_BIG;
      float s1 = ok1 ? sc1[r] : NEG_BIG;
      float tmax = half16_reduce_max(fmaxf(s0, s1));
      float mnew = fmaxf(mrow[r], tmax);
      float corr = __expf(mrow[r] - mnew);
      float p0 = ok0 ? __expf(s0 - mnew) : 0.f;
      float p1 = ok1 ? __expf(s1 - mnew) : 0.f;
      lrow[r] = lrow[r] * corr + half16_reduce_sum(p0 + p1);
      mrow[r] = mnew;
      #pragma unroll
      for (int cd = 0; cd < 8; ++cd) Oacc[cd][r] *= corr;
      ps[(mbase + r) * 33 + ln]      = p0;
      ps[(mbase + r) * 33 + 16 + ln] = p1;
    }
    asm volatile("" ::: "memory");
    v16bf pA;
    {
      const int ka = hi ? 8 : 0;
      const float* pr = ps + (size_t)ln * 33;
      #pragma unroll
      for (int e = 0; e < 8; ++e) {
        pA[e]     = (__bf16)pr[2 * ka + e];
        pA[8 + e] = (__bf16)pr[2 * ka + 8 + e];
      }
    }
    #pragma unroll
    for (int cd = 0; cd < 8; ++cd) {
      v16bf bV;
      const float* vr = vb + (size_t)(kt + kbB) * D_HEAD + cd * 16 + ln;
      #pragma unroll
      for (int e = 0; e < 16; ++e) bV[e] = (__bf16)vr[(size_t)e * D_HEAD];
      Oacc[cd] = __builtin_amdgcn_wmma_f32_16x16x32_bf16(false, pA, false, bV,
                                                         (short)0, Oacc[cd], false, false);
    }
  }
  #pragma unroll
  for (int r = 0; r < 8; ++r) {
    const int row = m0 + mbase + r;
    const float inv = 1.0f / lrow[r];
    float* orow = Out + ((size_t)b * S_LEN + row) * (N_HEADS * D_HEAD) + h * D_HEAD + ln;
    #pragma unroll
    for (int cd = 0; cd < 8; ++cd) orow[cd * 16] = Oacc[cd][r] * inv;
  }
}

extern "C" void kernel_launch(void* const* d_in, const int* in_sizes, int n_in,
                              void* d_out, int out_size, void* d_ws, size_t ws_size,
                              hipStream_t stream) {
  (void)in_sizes; (void)n_in; (void)out_size;
  const float* q = (const float*)d_in[0];
  const float* k = (const float*)d_in[1];
  const float* v = (const float*)d_in[2];
  const int* layer = (const int*)d_in[3];
  float* out = (float*)d_out;

  const size_t elems  = (size_t)N_BATCH * N_HEADS * S_LEN * D_HEAD;
  const size_t need   = 3 * elems * sizeof(__bf16);
  const int    blocks = N_BATCH * N_HEADS * (S_LEN / QTILE) / 4;   // 1024

  if (ws_size >= need) {
    __bf16* qb = (__bf16*)d_ws;
    __bf16* kb = qb + elems;
    __bf16* vt = kb + elems;
    convert_qk<<<2048, 256, 0, stream>>>(q, k, qb, kb);
    transpose_v<<<dim3(S_LEN / 32, N_BATCH * N_HEADS), 256, 0, stream>>>(v, vt);
    fattn_bf16<<<blocks, 128, 0, stream>>>(qb, kb, vt, layer, out);
  } else {
    fattn_fallback<<<blocks, 128, 0, stream>>>(q, k, v, layer, out);
  }
}
